// GCNN_layer_59622736003846
// MI455X (gfx1250) — matrix operand order repaired and checked by reading, not compiled, not run on hardware
//
#include <hip/hip_runtime.h>
#include <hip/hip_bf16.h>
#include <math.h>

typedef __attribute__((ext_vector_type(16))) _Float16 v16h;
typedef __attribute__((ext_vector_type(8)))  _Float16 v8h;
typedef __attribute__((ext_vector_type(8)))  float    v8f;
typedef __attribute__((ext_vector_type(2)))  int      v2i;

#define B_   16
#define CIN  128
#define COUT 128
#define H_   64
#define W_   64
#define KS   7
#define PAD  3
#define G_   4
#define HP   (H_ + 2*PAD)   // 70
#define WP2  72             // padded row width (2 extra zero cols -> 18432B rows)
#define MTOT (G_*COUT)      // 512
#define KTOT (CIN*KS*KS)    // 6272
#define NKT  (KTOT/32)      // 196 K-tiles of 32
#define LROW 136            // LDS row stride in f16 (272B = 68 DW -> conflict-free)
#define LBUF (WP2*LROW)     // 9792 f16 per buffer

// Detect gfx1250 async global->LDS path
#if defined(__has_builtin)
#if __has_builtin(__builtin_amdgcn_global_load_async_to_lds_b64) && \
    __has_builtin(__builtin_amdgcn_s_wait_asynccnt)
#define USE_ASYNC 1
#endif
#endif
#ifndef USE_ASYNC
#define USE_ASYNC 0
#endif

#if USE_ASYNC
typedef __attribute__((address_space(1))) v2i* as1_v2i;
typedef __attribute__((address_space(3))) v2i* as3_v2i;
#endif

// ---------------------------------------------------------------------------
// 1) NCHW f32 -> padded NHWC f16 (zero borders), layout [B][HP][WP2][CIN]
// ---------------------------------------------------------------------------
__global__ __launch_bounds__(256) void pad_cvt_kernel(const float* __restrict__ x,
                                                      _Float16* __restrict__ xpad) {
    int gt  = blockIdx.x * 256 + threadIdx.x;     // 16*70*72*16 tasks
    int wp  = gt % WP2; int t = gt / WP2;
    int hp  = t % HP;   t /= HP;
    int cig = t % 16;   int b = t / 16;           // cig: group of 8 channels
    int h = hp - PAD, w = wp - PAD;
    v8h v;
    if (h >= 0 && h < H_ && w >= 0 && w < W_) {
        const float* src = x + (((size_t)b*CIN + cig*8)*H_ + h)*W_ + w;
#pragma unroll
        for (int j = 0; j < 8; ++j) v[j] = (_Float16)src[(size_t)j*H_*W_];
    } else {
#pragma unroll
        for (int j = 0; j < 8; ++j) v[j] = (_Float16)0.f;
    }
    *(v8h*)(xpad + ((size_t)(b*HP + hp)*WP2 + wp)*CIN + cig*8) = v;
}

// ---------------------------------------------------------------------------
// 2) Generate 4 rotated filter banks, write f16 pre-swizzled for WMMA A-operand
//    A layout (16-bit 16x32): lane = khalf*16 + m; slot s<8 -> K=khalf*8+s,
//    s>=8 -> K=16+khalf*8+(s-8). Stored as [cot][kt][lane][16] f16.
// ---------------------------------------------------------------------------
__device__ __forceinline__ float sinc_n(float z) {
    float pz = 3.14159265358979323846f * z;
    if (fabsf(z) < 1e-7f) return 1.0f;
    return sinf(pz) / pz;
}

__global__ __launch_bounds__(256) void gen_filters_kernel(const float* __restrict__ wgt,
                                                          _Float16* __restrict__ aswz) {
    __shared__ float Msh[49*49];
    float c = cosf(90.0f), s = sinf(90.0f);   // THETA = 90 *radians*, faithful
    for (int idx = threadIdx.x; idx < 49*49; idx += 256) {
        int q = idx / 49, p = idx % 49;
        int a  = q / 7, bq = q % 7;
        int pa = p / 7, pb = p % 7;
        float cxa = (float)(pa - 3), cxb = (float)(pb - 3);
        float tx = c*cxa - s*cxb + 2.0f;      // +k//2-1 (asymmetric, as in torch)
        float ty = s*cxa + c*cxb + 3.0f;      // +k//2
        Msh[idx] = sinc_n(tx - (float)a) * sinc_n(ty - (float)bq);
    }
    __syncthreads();

    int oi = blockIdx.x * 256 + threadIdx.x;  // (o, ci) pair, 16384 total
    int o = oi >> 7, ci = oi & 127;
    float buf[49], nbuf[49];
    for (int p = 0; p < 49; ++p) buf[p] = wgt[(size_t)oi*49 + p];

    for (int g = 0; g < G_; ++g) {
        int mg  = g*COUT + o;                 // row in the 512-wide M dim
        int cot = mg >> 4, m = mg & 15;
        for (int p = 0; p < 49; ++p) {
            int k  = p*CIN + ci;              // K = (kh*7+kw)*128 + ci
            int kt = k >> 5, kk = k & 31;
            int khalf = (kk >> 3) & 1;
            int slot  = (kk & 7) + ((kk & 16) ? 8 : 0);
            int lane  = khalf*16 + m;
            aswz[(((size_t)cot*NKT + kt)*32 + lane)*16 + slot] = (_Float16)buf[p];
        }
        if (g < G_ - 1) {
            for (int p = 0; p < 49; ++p) {
                float acc = 0.f;
                for (int q = 0; q < 49; ++q) acc += buf[q] * Msh[q*49 + p];
                nbuf[p] = acc;
            }
            for (int p = 0; p < 49; ++p) buf[p] = nbuf[p];
        }
    }
}

// ---------------------------------------------------------------------------
// Stage one padded input row (b, h+kh): 72*128 f16 = 18432B = 9 x B64 / thread.
// Async (ASYNCcnt) global->LDS on gfx1250, sync fallback otherwise.
// ---------------------------------------------------------------------------
__device__ __forceinline__ void stage_slice(const _Float16* __restrict__ g,
                                            _Float16* l, int tid) {
#pragma unroll
    for (int i = 0; i < 9; ++i) {
        int c = tid + i*256;              // 0..2303 B64 chunks
        int w = c >> 5, part = c & 31;    // 32 chunks per 256B row
        int go = w*CIN  + part*4;         // f16 units
        int lo = w*LROW + part*4;
#if USE_ASYNC
        __builtin_amdgcn_global_load_async_to_lds_b64(
            (as1_v2i)(g + go), (as3_v2i)(l + lo), 0, 0);
#else
        *(unsigned long long*)(l + lo) = *(const unsigned long long*)(g + go);
#endif
    }
}

// ---------------------------------------------------------------------------
// 3) Implicit-GEMM conv via v_wmma_f32_16x16x32_f16, LDS-staged B operand.
//    M=512 (co), N=65536 (b,h,w), K=6272. Block: 8 waves = 4(M) x 2(N),
//    covers 128 co x 64 px (one b,h row); wave: 2x2 tiles of 16x16.
// ---------------------------------------------------------------------------
__global__ __launch_bounds__(256) void conv_wmma_kernel(const _Float16* __restrict__ xpad,
                                                        const _Float16* __restrict__ aswz,
                                                        float* __restrict__ out) {
    __shared__ _Float16 smem[2*LBUF];     // double-buffered row slice (39168B)

    int tid   = threadIdx.x;
    int lane  = tid & 31;
    int wv    = tid >> 5;
    int mwave = wv & 3;          // 4 waves along M
    int nwave = wv >> 2;         // 2 waves along N
    int mb    = blockIdx.x & 3;  // 4 co blocks of 128
    int bh    = blockIdx.x >> 2;
    int b     = bh >> 6;
    int h     = bh & 63;

    int cot0  = mb*8 + mwave*2;  // M-tile index (of 32)
    int wbase = nwave*32;

    int nlo = lane & 15;         // pixel within 16-wide N tile
    int kbo = (lane >> 4) * 8;   // B-operand: ci offset for this half-wave

    v8f acc00 = {}, acc01 = {}, acc10 = {}, acc11 = {};

    const _Float16* a0p = aswz + ((size_t)cot0       * NKT * 32 + lane) * 16;
    const _Float16* a1p = aswz + ((size_t)(cot0 + 1) * NKT * 32 + lane) * 16;
    const _Float16* xbase = xpad + (size_t)(b*HP + h) * (WP2*CIN);

    stage_slice(xbase, smem, tid);                        // kh = 0 into buf0

    for (int kh = 0; kh < KS; ++kh) {
        _Float16* cur = smem + (kh & 1) * LBUF;
        if (kh < KS-1)
            stage_slice(xbase + (size_t)(kh+1)*(WP2*CIN),
                        smem + ((kh+1) & 1) * LBUF, tid);
#if USE_ASYNC
        if (kh < KS-1) __builtin_amdgcn_s_wait_asynccnt(9);  // prev stage done
        else           __builtin_amdgcn_s_wait_asynccnt(0);
#endif
        __syncthreads();

        int ktb = kh * 28;
        for (int kw = 0; kw < KS; ++kw) {
            const _Float16* lrow = cur + (wbase + nlo + kw) * LROW + kbo;
#pragma unroll
            for (int cit = 0; cit < 4; ++cit) {
                int kt = ktb + kw*4 + cit;
                v16h a0 = *(const v16h*)(a0p + (size_t)kt * 32 * 16);
                v16h a1 = *(const v16h*)(a1p + (size_t)kt * 32 * 16);

                const _Float16* lb0 = lrow + cit*32;
                const _Float16* lb1 = lb0 + 16*LROW;
                v8h l0 = *(const v8h*)lb0, h0 = *(const v8h*)(lb0 + 16);
                v8h l1 = *(const v8h*)lb1, h1 = *(const v8h*)(lb1 + 16);
                v16h b0 = __builtin_shufflevector(l0, h0, 0,1,2,3,4,5,6,7,8,9,10,11,12,13,14,15);
                v16h b1 = __builtin_shufflevector(l1, h1, 0,1,2,3,4,5,6,7,8,9,10,11,12,13,14,15);

                acc00 = __builtin_amdgcn_wmma_f32_16x16x32_f16(false, a0, false, b0,
                                                               (short)0, acc00, false, false);
                acc01 = __builtin_amdgcn_wmma_f32_16x16x32_f16(false, a0, false, b1,
                                                               (short)0, acc01, false, false);
                acc10 = __builtin_amdgcn_wmma_f32_16x16x32_f16(false, a1, false, b0,
                                                               (short)0, acc10, false, false);
                acc11 = __builtin_amdgcn_wmma_f32_16x16x32_f16(false, a1, false, b1,
                                                               (short)0, acc11, false, false);
            }
        }
        if (kh < KS-1) __syncthreads();   // protect buffer reuse two stages later
    }

    // C/D layout: lanes 0-15: N=lane, VGPR r -> M=r; lanes 16-31: M=8+r, N=lane-16
    int mrow   = (lane >> 4) * 8;
    int ncol   = lane & 15;
    int cobase = mb*128 + mwave*32;
    size_t outb = (size_t)b * MTOT * (H_*W_) + (size_t)h * W_;
#pragma unroll
    for (int r = 0; r < 8; ++r) {
        out[outb + (size_t)(cobase      + mrow + r) * (H_*W_) + wbase      + ncol] = acc00[r];
        out[outb + (size_t)(cobase      + mrow + r) * (H_*W_) + wbase + 16 + ncol] = acc01[r];
        out[outb + (size_t)(cobase + 16 + mrow + r) * (H_*W_) + wbase      + ncol] = acc10[r];
        out[outb + (size_t)(cobase + 16 + mrow + r) * (H_*W_) + wbase + 16 + ncol] = acc11[r];
    }
}

// ---------------------------------------------------------------------------
extern "C" void kernel_launch(void* const* d_in, const int* in_sizes, int n_in,
                              void* d_out, int out_size, void* d_ws, size_t ws_size,
                              hipStream_t stream) {
    (void)in_sizes; (void)n_in; (void)out_size; (void)ws_size;
    const float* x   = (const float*)d_in[0];
    const float* wgt = (const float*)d_in[1];
    float* out = (float*)d_out;
    char*  ws  = (char*)d_ws;

    _Float16* xpad = (_Float16*)ws;                         // 16*70*72*128 f16 = 20,643,840 B
    size_t xpad_bytes = (size_t)B_ * HP * WP2 * CIN * sizeof(_Float16);
    _Float16* aswz = (_Float16*)(ws + xpad_bytes);          // 32*196*32*16 f16 = 6,422,528 B

    pad_cvt_kernel<<<(B_*HP*WP2*16)/256, 256, 0, stream>>>(x, xpad);
    gen_filters_kernel<<<(COUT*CIN)/256, 256, 0, stream>>>(wgt, aswz);
    conv_wmma_kernel<<<4 * B_ * H_, 256, 0, stream>>>(xpad, aswz, out);
}
